// SimpleRNN_87686052315342
// MI455X (gfx1250) — compile-verified, matrix operand order checked by reading, not run on hardware
//
#include <hip/hip_runtime.h>

// ---------------------------------------------------------------------------
// MI455X / gfx1250 LSTM: bf16 WMMA (16x16x32), fp32 accumulate.
//
// v3: - input projection folded into layer 0: Wf = W_ih0 @ W_in (fp32
//       preprocessing GEMM), bias b0f = b_ih0 + b_hh0 + W_ih0 @ b_in.
//       Layer-0 A operand streams straight from fp32 `inputs`.
//     - batch tile M=32 per workgroup; each packed B fragment (weights)
//       feeds two WMMAs (row tiles) -> half the L2 weight traffic of M=16.
//     - register pressure fix vs v2 (which spilled): quarter K-sweeps
//       (8 f32 accumulators), cell state in LDS (fp32), ping-pong h buffers
//       in LDS so epilogues write new h directly. ~230 live VGPRs.
// ---------------------------------------------------------------------------

typedef __bf16 bf16;
typedef __attribute__((ext_vector_type(16))) __bf16 v16bf;
typedef __attribute__((ext_vector_type(8)))  __bf16 v8bf;
typedef __attribute__((ext_vector_type(8)))  float  v8f;

#define Bc 4096
#define Tc 10
#define Hc 512
#define Dout 51
#define Mrows 32                                  // batch rows per workgroup
#define OFF_H ((size_t)Bc * Dout)                 // start of hN in d_out
#define OFF_C (OFF_H + (size_t)2 * Bc * Hc)       // start of cN in d_out

__device__ __forceinline__ float sigm(float x) { return 1.0f / (1.0f + __expf(-x)); }
__device__ __forceinline__ float tanh_f(float x) { return 1.0f - 2.0f / (__expf(2.0f * x) + 1.0f); }

__device__ __forceinline__ v8f wmma_bf16(v16bf a, v16bf b, v8f c) {
  return __builtin_amdgcn_wmma_f32_16x16x32_bf16(false, a, false, b, (short)0, c,
                                                 false, false);
}

// ISA 16-bit A-fragment K map: lanes<16: K = e (e<8), e+8 (e>=8);
// lanes>=16: +8. Two contiguous 16B chunks per lane.
__device__ __forceinline__ v16bf load_a_lds(const bf16* base, int k0, int lr, int lh) {
  const bf16* p = base + lr * Hc + k0 + 8 * lh;
  v8bf lo = *(const v8bf*)p;
  v8bf hi = *(const v8bf*)(p + 16);
  v16bf a;
#pragma unroll
  for (int i = 0; i < 8; ++i) { a[i] = lo[i]; a[i + 8] = hi[i]; }
  return a;
}

// Same A fragment, sourced from fp32 row-major global (converted on the fly).
__device__ __forceinline__ v16bf load_a_f32(const float* rowptr, int k0, int lh) {
  const float* p = rowptr + k0 + 8 * lh;
  v16bf a;
#pragma unroll
  for (int i = 0; i < 8; ++i) { a[i] = (bf16)p[i]; a[i + 8] = (bf16)p[i + 16]; }
  return a;
}

// ---------------------------------------------------------------------------
// Preprocessing: Wf = W_ih0 @ W_in  (2048x512 = 2048x512 @ 512x512, fp32).
// One block per output row n; 256 threads cover k and k+256.
// ---------------------------------------------------------------------------
__global__ void fuse_w(const float* __restrict__ Wih0, const float* __restrict__ Win,
                       float* __restrict__ Wf) {
  const int n = blockIdx.x;
  const int k = threadIdx.x;               // k and k+256
  float a0 = 0.f, a1 = 0.f;
  for (int m = 0; m < Hc; ++m) {
    const float wm = Wih0[(size_t)n * Hc + m];
    a0 += wm * Win[(size_t)m * Hc + k];
    a1 += wm * Win[(size_t)m * Hc + k + 256];
  }
  Wf[(size_t)n * Hc + k] = a0;
  Wf[(size_t)n * Hc + k + 256] = a1;
}

// b0f = b_ih0 + b_hh0 + W_ih0 @ b_in ;  b1f = b_ih1 + b_hh1
__global__ void fuse_b(const float* __restrict__ Wih0, const float* __restrict__ b_in,
                       const float* __restrict__ bih0, const float* __restrict__ bhh0,
                       const float* __restrict__ bih1, const float* __restrict__ bhh1,
                       float* __restrict__ b0f, float* __restrict__ b1f) {
  const int n = blockIdx.x * blockDim.x + threadIdx.x;
  if (n < 4 * Hc) {
    float s = bih0[n] + bhh0[n];
    for (int m = 0; m < Hc; ++m) s += Wih0[(size_t)n * Hc + m] * b_in[m];
    b0f[n] = s;
    b1f[n] = bih1[n] + bhh1[n];
  }
}

// ---------------------------------------------------------------------------
// Weight packing: W is (N x K) fp32 row-major; gates = x @ W^T so B[k][n] =
// W[n][k]. Pack per (nb,kb) 32x16 block into WMMA B lane order: lane l holds
// 16 contiguous bf16. Block index = nb*KB + kb.
// ---------------------------------------------------------------------------
__global__ void pack_w(const float* __restrict__ W, bf16* __restrict__ P,
                       int K, int Nact) {
  const int KB = K >> 5;
  const int nb = blockIdx.x / KB, kb = blockIdx.x % KB;
  const int l = threadIdx.x;                  // 0..31
  const int n = nb * 16 + (l & 15);
  const int kbase = kb * 32 + 8 * (l >> 4);
  bf16* dst = P + (size_t)blockIdx.x * 512 + l * 16;
#pragma unroll
  for (int e = 0; e < 16; ++e) {
    const int k = kbase + (e < 8 ? e : e + 8);
    const float v = (n < Nact) ? W[(size_t)n * K + k] : 0.0f;
    dst[e] = (bf16)v;
  }
}

// ---------------------------------------------------------------------------
// One LSTM layer step for a 32-row batch tile. Wave w owns h-columns
// [w*64, w*64+64) and all four gates for them. Four K-sweeps, one 16-col
// subtile each (8 accumulators). B fragments feed 2 WMMAs (row tiles).
// New h written directly to the ping-pong hnxt buffer; cell state in LDS.
// ---------------------------------------------------------------------------
__device__ __forceinline__ void lstm_layer(
    bool xFromGlobal,
    const float* xrow0, const float* xrow1,   // per-lane fp32 rows (layer 0)
    const bf16* xlds,                         // 32x512 LDS x source (layer 1)
    const bf16* hcur, bf16* hnxt,             // 32x512 LDS ping-pong h
    float* cls,                               // 32x512 LDS fp32 cell state
    const v16bf* __restrict__ Pih, const v16bf* __restrict__ Phh,
    const float* bb,                          // [16] per-lane biases g*4+j
    int w, int l, int lr, int lh,
    bool writeHC, float* __restrict__ gH, float* __restrict__ gC, int row0) {
#pragma unroll
  for (int j = 0; j < 4; ++j) {               // 16-col subtile sweep
    v8f acc[8] = {};                          // idx rt*4 + g
    for (int kb = 0; kb < 16; ++kb) {         // K = 512 in steps of 32
      const int k0 = kb * 32;
      v16bf ax0 = xFromGlobal ? load_a_f32(xrow0, k0, lh)
                              : load_a_lds(xlds, k0, lr, lh);
      v16bf ax1 = xFromGlobal ? load_a_f32(xrow1, k0, lh)
                              : load_a_lds(xlds + 16 * Hc, k0, lr, lh);
      v16bf ah0 = load_a_lds(hcur, k0, lr, lh);
      v16bf ah1 = load_a_lds(hcur + 16 * Hc, k0, lr, lh);
#pragma unroll
      for (int g = 0; g < 4; ++g) {
        const int nb = g * 32 + w * 4 + j;
        const size_t blk = (size_t)(nb * 16 + kb) * 32 + l;
        const v16bf bi = Pih[blk];
        const v16bf bh = Phh[blk];
        acc[g]     = wmma_bf16(ax0, bi, acc[g]);
        acc[g]     = wmma_bf16(ah0, bh, acc[g]);
        acc[4 + g] = wmma_bf16(ax1, bi, acc[4 + g]);
        acc[4 + g] = wmma_bf16(ah1, bh, acc[4 + g]);
      }
    }
    const int col = w * 64 + j * 16 + lr;     // C/D: N = lane&15
#pragma unroll
    for (int rt = 0; rt < 2; ++rt) {
      v8f gi = acc[rt * 4 + 0], gf = acc[rt * 4 + 1];
      v8f gg = acc[rt * 4 + 2], go = acc[rt * 4 + 3];
#pragma unroll
      for (int e = 0; e < 8; ++e) {           // C/D: M = e + 8*(lane>=16)
        const int m = rt * 16 + e + 8 * lh;
        const float iv = sigm(gi[e] + bb[j]);
        const float fv = sigm(gf[e] + bb[4 + j]);
        const float gv = tanh_f(gg[e] + bb[8 + j]);
        const float ov = sigm(go[e] + bb[12 + j]);
        const float cn = fv * cls[m * Hc + col] + iv * gv;
        const float hn = ov * tanh_f(cn);
        cls[m * Hc + col] = cn;
        hnxt[m * Hc + col] = (bf16)hn;
        if (writeHC) {
          gH[(size_t)(row0 + m) * Hc + col] = hn;
          gC[(size_t)(row0 + m) * Hc + col] = cn;
        }
      }
    }
  }
}

__global__ void __launch_bounds__(256) lstm_kernel(
    const float* __restrict__ inputs,
    const v16bf* __restrict__ Pfuse, const v16bf* __restrict__ Phh0,
    const float* __restrict__ b0f,
    const v16bf* __restrict__ Pih1, const v16bf* __restrict__ Phh1,
    const float* __restrict__ b1f,
    float* __restrict__ gH, float* __restrict__ gC) {
  __shared__ bf16 h0a[Mrows * Hc], h0b[Mrows * Hc];
  __shared__ bf16 h1a[Mrows * Hc], h1b[Mrows * Hc];
  __shared__ float c0ls[Mrows * Hc], c1ls[Mrows * Hc];
  const int tid = threadIdx.x, w = tid >> 5, l = tid & 31, lr = l & 15, lh = l >> 4;
  const int row0 = blockIdx.x * Mrows;

  for (int i = tid; i < Mrows * Hc; i += 256) {
    h0a[i] = (bf16)0.0f; h1a[i] = (bf16)0.0f;
    c0ls[i] = 0.0f; c1ls[i] = 0.0f;
  }

  // Per-lane bias preload (constant across timesteps).
  float bb0[16], bb1[16];
#pragma unroll
  for (int j = 0; j < 4; ++j) {
    const int col = w * 64 + j * 16 + lr;
#pragma unroll
    for (int g = 0; g < 4; ++g) {
      bb0[g * 4 + j] = b0f[g * Hc + col];
      bb1[g * 4 + j] = b1f[g * Hc + col];
    }
  }
  __syncthreads();

  for (int t = 0; t < Tc; ++t) {
    bf16* h0c = (t & 1) ? h0b : h0a;  bf16* h0n = (t & 1) ? h0a : h0b;
    bf16* h1c = (t & 1) ? h1b : h1a;  bf16* h1n = (t & 1) ? h1a : h1b;
    const float* xr0 = inputs + ((size_t)(row0 + lr) * Tc + t) * Hc;
    const float* xr1 = inputs + ((size_t)(row0 + 16 + lr) * Tc + t) * Hc;
    const bool last = (t == Tc - 1);

    // Layer 0: gates = x_t @ Wf^T + h0 @ W_hh0^T + b0f
    lstm_layer(true, xr0, xr1, nullptr, h0c, h0n, c0ls, Pfuse, Phh0, bb0,
               w, l, lr, lh, last, gH, gC, row0);
    __syncthreads();                   // new h0 visible; old h0 reads done
    // Layer 1: gates = h0_new @ W_ih1^T + h1 @ W_hh1^T + b1f
    lstm_layer(false, nullptr, nullptr, h0n, h1c, h1n, c1ls, Pih1, Phh1, bb1,
               w, l, lr, lh, last,
               gH + (size_t)Bc * Hc, gC + (size_t)Bc * Hc, row0);
    __syncthreads();                   // new h1 visible; old h1 reads done
  }
}

// out = h1_final @ W_out^T + b_out ; N padded 51 -> 64, masked store.
__global__ void __launch_bounds__(256) outproj_kernel(
    const float* __restrict__ h1, const v16bf* __restrict__ Pout,
    const float* __restrict__ b_out, float* __restrict__ out) {
  const int tid = threadIdx.x, w = tid >> 5, l = tid & 31, lr = l & 15, lh = l >> 4;
  const int row0 = (blockIdx.x * 8 + w) * 16;
  const float* arow = h1 + (size_t)(row0 + lr) * Hc;
  v8f acc[4] = {};
  for (int kb = 0; kb < 16; ++kb) {
    v16bf a = load_a_f32(arow, kb * 32, lh);
#pragma unroll
    for (int j = 0; j < 4; ++j)
      acc[j] = wmma_bf16(a, Pout[(size_t)(j * 16 + kb) * 32 + l], acc[j]);
  }
#pragma unroll
  for (int j = 0; j < 4; ++j) {
    const int n = j * 16 + lr;
    if (n < Dout) {
      const float bb = b_out[n];
#pragma unroll
      for (int e = 0; e < 8; ++e) {
        const int m = e + 8 * lh;
        out[(size_t)(row0 + m) * Dout + n] = acc[j][e] + bb;
      }
    }
  }
}

extern "C" void kernel_launch(void* const* d_in, const int* in_sizes, int n_in,
                              void* d_out, int out_size, void* d_ws, size_t ws_size,
                              hipStream_t stream) {
  (void)in_sizes; (void)n_in; (void)out_size; (void)ws_size;
  const float* inputs = (const float*)d_in[0];
  const float* W_in  = (const float*)d_in[1];
  const float* b_in  = (const float*)d_in[2];
  const float* W_ih0 = (const float*)d_in[3];
  const float* W_hh0 = (const float*)d_in[4];
  const float* b_ih0 = (const float*)d_in[5];
  const float* b_hh0 = (const float*)d_in[6];
  const float* W_ih1 = (const float*)d_in[7];
  const float* W_hh1 = (const float*)d_in[8];
  const float* b_ih1 = (const float*)d_in[9];
  const float* b_hh1 = (const float*)d_in[10];
  const float* W_out = (const float*)d_in[11];
  const float* b_out = (const float*)d_in[12];
  float* out = (float*)d_out;

  // Workspace layout (~12.7 MB total).
  char* ws = (char*)d_ws;
  size_t o = 0;
  float* Wf   = (float*)(ws + o); o += (size_t)2048 * 512 * 4;  // fused fp32
  bf16* Pfuse = (bf16*)(ws + o);  o += (size_t)2048 * 512 * 2;
  bf16* Phh0  = (bf16*)(ws + o);  o += (size_t)2048 * 512 * 2;
  bf16* Pih1  = (bf16*)(ws + o);  o += (size_t)2048 * 512 * 2;
  bf16* Phh1  = (bf16*)(ws + o);  o += (size_t)2048 * 512 * 2;
  bf16* Pout  = (bf16*)(ws + o);  o += (size_t)64 * 512 * 2;    // N pad -> 64
  float* b0f  = (float*)(ws + o); o += (size_t)2048 * 4;
  float* b1f  = (float*)(ws + o); o += (size_t)2048 * 4;

  // 1) Fuse input projection into layer 0 (fp32).
  fuse_w<<<2048, 256, 0, stream>>>(W_ih0, W_in, Wf);
  fuse_b<<<8, 256, 0, stream>>>(W_ih0, b_in, b_ih0, b_hh0, b_ih1, b_hh1, b0f, b1f);

  // 2) Pack weights to bf16 WMMA-fragment order.
  pack_w<<<dim3(128 * 16), 32, 0, stream>>>(Wf,    Pfuse, 512, 2048);
  pack_w<<<dim3(128 * 16), 32, 0, stream>>>(W_hh0, Phh0,  512, 2048);
  pack_w<<<dim3(128 * 16), 32, 0, stream>>>(W_ih1, Pih1,  512, 2048);
  pack_w<<<dim3(128 * 16), 32, 0, stream>>>(W_hh1, Phh1,  512, 2048);
  pack_w<<<dim3(4 * 16), 32, 0, stream>>>(W_out,  Pout,  512, 51);

  // 3) Recurrent sweep: writes hN at out+OFF_H, cN at out+OFF_C.
  lstm_kernel<<<Bc / Mrows, 256, 0, stream>>>(
      inputs,
      (const v16bf*)Pfuse, (const v16bf*)Phh0, b0f,
      (const v16bf*)Pih1, (const v16bf*)Phh1, b1f,
      out + OFF_H, out + OFF_C);

  // 4) Readout from final h1 (fp32, already in d_out's hN[1] region).
  outproj_kernel<<<Bc / 16 / 8, 256, 0, stream>>>(
      out + OFF_H + (size_t)Bc * Hc, (const v16bf*)Pout, b_out, out);
}
